// AstronomicalStructureLoss_5334349382000
// MI455X (gfx1250) — matrix operand order repaired and checked by reading, not compiled, not run on hardware
//
#include <hip/hip_runtime.h>
#include <math.h>

// ---------------- CDNA5 types ----------------
typedef __attribute__((ext_vector_type(16))) __bf16 v16bf;
typedef __attribute__((ext_vector_type(8)))  float  v8f;
typedef unsigned int u32x4 __attribute__((ext_vector_type(4)));
typedef int          i32x8 __attribute__((ext_vector_type(8)));
typedef int          i32x4 __attribute__((ext_vector_type(4)));

#define IMG      512
#define NIMG     64
#define TILE     64
#define HALO     3
#define DDIM     (TILE + 2*HALO)   // 70: staged tile (halo 3), compact stride
#define NTOT     16777216.0f       // 64*512*512
#define NEGINF   (-3.0e38f)

// Tensor Data Mover available? (builtin exists on both probed toolchains,
// with different arity: ROCm clang-22 = 5 args, upstream clang-23 = 6 args)
#if defined(__has_builtin)
#  if __has_builtin(__builtin_amdgcn_tensor_load_to_lds) && \
      __has_builtin(__builtin_amdgcn_s_wait_tensorcnt)
#    define USE_TDM 1
#  endif
#endif
#ifndef USE_TDM
#  define USE_TDM 0
#endif

union AFrag { v16bf v; unsigned short u[16]; };

// round-to-nearest-even fp32 -> bf16 bits
__device__ __forceinline__ unsigned f2bf(float f) {
  union { float f; unsigned u; } c; c.f = f;
  unsigned u = c.u;
  u += 0x7FFFu + ((u >> 16) & 1u);
  return u >> 16;
}

// Filter bank: 9 channels in 7x7 support, tap k -> (dy,dx)=(k/7-3,k%7-3)
// m=0 SobelX, 1 SobelY, 2 Laplacian, 3..5 Gauss(0.5,1,2), 6..8 LoG(1,2,4)
// grid: x_i = -4 + i*(7/6)   (torch.linspace(-4,3,7))
__device__ float raw_tap(int m, int k) {
  if (m > 8 || k >= 49) return 0.0f;
  int q  = k / 7;
  int dy = q - 3, dx = (k - q * 7) - 3;
  if (m < 3) {
    if (dy < -1 || dy > 1 || dx < -1 || dx > 1) return 0.0f;
    if (m == 0) return (float)dx * ((dy == 0) ? 2.0f : 1.0f);
    if (m == 1) return (float)dy * ((dx == 0) ? 2.0f : 1.0f);
    int ad = (dy < 0 ? -dy : dy) + (dx < 0 ? -dx : dx);
    return (ad == 0) ? 4.0f : ((ad == 1) ? -1.0f : 0.0f);
  }
  float sig = (m == 3) ? 0.5f : (m == 4) ? 1.0f : (m == 5) ? 2.0f
            : (m == 6) ? 1.0f : (m == 7) ? 2.0f : 4.0f;
  float X = -4.0f + (float)(dy + 3) * (7.0f / 6.0f);
  float Y = -4.0f + (float)(dx + 3) * (7.0f / 6.0f);
  float r2 = X * X + Y * Y;
  float s2 = 2.0f * sig * sig;
  float e  = expf(-r2 / s2);
  return (m <= 5) ? e : (1.0f - r2 / s2) * e;
}

#if USE_TDM
// Issue one TDM 2D tile load: 70x70 fp32 tile of a 512-stride image -> LDS.
__device__ __forceinline__ void tdm_load_tile(const float* gsrc, unsigned lds_off) {
  unsigned long long ga = (unsigned long long)(size_t)gsrc;
  u32x4 g0;
  g0[0] = 1u;                                      // count=1, user descriptor
  g0[1] = lds_off;                                 // lds_addr (bytes)
  g0[2] = (unsigned)ga;                            // global_addr[31:0]
  g0[3] = (unsigned)((ga >> 32) & 0x01FFFFFFu)     // global_addr[56:32]
        | 0x80000000u;                             // type = 2 ("image")
  i32x8 g1;
  g1[0] = (int)(2u << 16);                         // data_size = 2 (4 bytes)
  g1[1] = (int)((unsigned)IMG << 16);              // tensor_dim0 = 512
  g1[2] = (int)((unsigned)IMG << 16);              // tensor_dim1 = 512
  g1[3] = (int)((unsigned)DDIM << 16);             // tile_dim0 = 70
  g1[4] = DDIM;                                    // tile_dim1 = 70
  g1[5] = IMG;                                     // tensor_dim0_stride = 512
  g1[6] = 0;
  g1[7] = 0;
  i32x4 g2 = {0, 0, 0, 0};                         // 2D tensor: groups 2/3 unused
  i32x4 g3 = {0, 0, 0, 0};
#if defined(__clang_major__) && __clang_major__ >= 23
  i32x8 g4 = {0, 0, 0, 0, 0, 0, 0, 0};
  __builtin_amdgcn_tensor_load_to_lds(g0, g1, g2, g3, g4, 0);
#else
  __builtin_amdgcn_tensor_load_to_lds(g0, g1, g2, g3, 0);
#endif
}
#endif

// ---------------- Kernel A: sum(target) for the weight threshold ----------------
__global__ void astro_tsum_kernel(const float* __restrict__ t, float* __restrict__ ws, int n4) {
  const float4* t4 = (const float4*)t;
  float s = 0.0f;
  for (int i = blockIdx.x * blockDim.x + threadIdx.x; i < n4; i += gridDim.x * blockDim.x) {
    float4 v = t4[i];
    s += (v.x + v.y) + (v.z + v.w);
  }
  for (int off = 16; off; off >>= 1) s += __shfl_xor(s, off);
  __shared__ float bsum[8];
  int lane = threadIdx.x & 31, wid = threadIdx.x >> 5;
  if (lane == 0) bsum[wid] = s;
  __syncthreads();
  if (threadIdx.x == 0) {
    float tot = 0.0f;
    for (int i = 0; i < (int)(blockDim.x >> 5); ++i) tot += bsum[i];
    atomicAdd(&ws[18], tot);
  }
}

// ---------------- Kernel B: fused loss ----------------
// ws[0..15]: per-filter-row sum of conv(d)^2 (rows 9..15 zero-filters)
// ws[16]: weighted recon SSE, ws[17]: sum|maxpool(a)-maxpool(b)|, ws[18]: sum(target)
__global__ void __launch_bounds__(256)
astro_loss_main(const float* __restrict__ A, const float* __restrict__ B,
                float* __restrict__ ws) {
  __shared__ float s_a[DDIM * DDIM];             // raw a, halo 3 (TDM-compatible layout)
  __shared__ float s_b[DDIM * DDIM];             // raw b, halo 3
  __shared__ unsigned short dlds[DDIM * DDIM];   // bf16(a-b), zero outside image
  __shared__ float red[18];

  const int tid = threadIdx.x;
  const int tr0 = blockIdx.y * TILE;
  const int tc0 = blockIdx.x * TILE;
  const size_t base = (size_t)blockIdx.z * (size_t)(IMG * IMG);
  const float* Ai = A + base;
  const float* Bi = B + base;

  if (tid < 18) red[tid] = 0.0f;

  const float tmean = ws[18] * (1.0f / NTOT);

  // Halo-3 region fully inside the image? (block-uniform)
  const bool interior = (blockIdx.x >= 1) && (blockIdx.x <= 6) &&
                        (blockIdx.y >= 1) && (blockIdx.y <= 6);

  // ---- Phase 1: stage raw a/b tiles into LDS ----
#if USE_TDM
  if (interior) {
    if (tid == 0) {  // single issue; TDM ignores EXEC, tracked by TENSORcnt
      const float* ga = Ai + (tr0 - HALO) * IMG + (tc0 - HALO);
      const float* gb = Bi + (tr0 - HALO) * IMG + (tc0 - HALO);
      tdm_load_tile(ga, (unsigned)(size_t)&s_a[0]);
      tdm_load_tile(gb, (unsigned)(size_t)&s_b[0]);
      __builtin_amdgcn_s_wait_tensorcnt(0);
    }
  } else
#endif
  {
    for (int idx = tid; idx < DDIM * DDIM; idx += 256) {
      int i = idx / DDIM, j = idx - i * DDIM;
      int gr = tr0 + i - HALO, gc = tc0 + j - HALO;
      bool ok = (gr >= 0) && (gr < IMG) && (gc >= 0) && (gc < IMG);
      int off = (ok ? gr : 0) * IMG + (ok ? gc : 0);
      s_a[idx] = ok ? Ai[off] : 0.0f;
      s_b[idx] = ok ? Bi[off] : 0.0f;
      if (ok && gr + 8 < IMG)
        __builtin_prefetch(Ai + off + 8 * IMG, 0, 0);   // global_prefetch_b8
    }
  }
  __syncthreads();

  // ---- Phase 2: build bf16 d-tile (packed pairs, LDS->LDS) ----
  for (int l = tid * 2; l < DDIM * DDIM; l += 512) {
    float2 av = *(const float2*)&s_a[l];
    float2 bv = *(const float2*)&s_b[l];
    unsigned lo = f2bf(av.x - bv.x);
    unsigned hi = f2bf(av.y - bv.y);
    ((unsigned*)dlds)[l >> 1] = lo | (hi << 16);
  }
  __syncthreads();

  // ---- Phase 2b: border tiles: out-of-image cells become -inf for max-pool ----
  if (!interior) {
    for (int idx = tid; idx < DDIM * DDIM; idx += 256) {
      int i = idx / DDIM, j = idx - i * DDIM;
      int gr = tr0 + i - HALO, gc = tc0 + j - HALO;
      bool ok = (gr >= 0) && (gr < IMG) && (gc >= 0) && (gc < IMG);
      if (!ok) { s_a[idx] = NEGINF; s_b[idx] = NEGINF; }
    }
  }
  __syncthreads();

  const int lane  = tid & 31;
  const int wid   = tid >> 5;
  const int n     = lane & 15;          // B/D column (pixel in group)
  const int khalf = (lane >> 4) * 16;   // B K-half: contiguous k = khalf + s, same order as A

  // ---- Constant A operand (filter bank) in the 16-bit 16x32 A layout ----
  const int m = n;                      // A row = filter channel (rows 9..15 zero)
  float norm = 1.0f;
  if (m >= 3 && m <= 8) {
    float s = 0.0f;
    for (int k = 0; k < 49; ++k) {
      float v = raw_tap(m, k);
      s += (m <= 5) ? v : fabsf(v);
    }
    norm = 1.0f / s;
  }
  AFrag a0, a1;
#pragma unroll
  for (int j = 0; j < 8; ++j) {
    int kk = 2 * j + ((lane >= 16) ? 16 : 0);
    a0.u[2 * j]     = (unsigned short)f2bf(raw_tap(m, kk) * norm);
    a0.u[2 * j + 1] = (unsigned short)f2bf(raw_tap(m, kk + 1) * norm);
    a1.u[2 * j]     = (unsigned short)f2bf(raw_tap(m, 32 + kk) * norm);
    a1.u[2 * j + 1] = (unsigned short)f2bf(raw_tap(m, 32 + kk + 1) * norm);
  }

  // ---- Phase 3: filter bank via v_wmma_f32_16x16x32_bf16 (2 per 16-px group) ----
  float accD[8];
#pragma unroll
  for (int e = 0; e < 8; ++e) accD[e] = 0.0f;

  for (int gi = wid; gi < TILE * (TILE / 16); gi += 8) {   // 256 groups, uniform/wave
    int r  = gi >> 2;
    int c0 = (gi & 3) << 4;
    int cb = c0 + n;
    AFrag b0, b1;
#pragma unroll
    for (int j = 0; j < 8; ++j) {
      int k0 = khalf + 2 * j;
      int qa = k0 / 7;        int ra = k0 - qa * 7;     // taps 0..31 in range
      int k1 = k0 + 1;        int qb = k1 / 7;  int rb = k1 - qb * 7;
      b0.u[2 * j]     = dlds[(r + qa) * DDIM + cb + ra];
      b0.u[2 * j + 1] = dlds[(r + qb) * DDIM + cb + rb];
      int k2 = k0 + 32;  bool v2 = k2 < 49;  int k2c = v2 ? k2 : 0;
      int qc = k2c / 7;  int rc = k2c - qc * 7;
      unsigned short t2 = dlds[(r + qc) * DDIM + cb + rc];
      int k3 = k0 + 33;  bool v3 = k3 < 49;  int k3c = v3 ? k3 : 0;
      int qd = k3c / 7;  int rd = k3c - qd * 7;
      unsigned short t3 = dlds[(r + qd) * DDIM + cb + rd];
      b1.u[2 * j]     = v2 ? t2 : (unsigned short)0;
      b1.u[2 * j + 1] = v3 ? t3 : (unsigned short)0;
    }
    v8f c = {0.f, 0.f, 0.f, 0.f, 0.f, 0.f, 0.f, 0.f};
    v8f y = __builtin_amdgcn_wmma_f32_16x16x32_bf16(false, a0.v, false, b0.v,
                                                    (short)0, c, false, false);
    y = __builtin_amdgcn_wmma_f32_16x16x32_bf16(false, a1.v, false, b1.v,
                                                (short)0, y, false, false);
#pragma unroll
    for (int e = 0; e < 8; ++e) { float t = y[e]; accD[e] += t * t; }
  }

  // ---- Phase 4: weighted recon MSE + max-pool L1 ----
  float accR = 0.0f, accP = 0.0f;
  for (int p = tid; p < TILE * TILE; p += 256) {
    int pr = p >> 6, pc = p & 63;
    int li = pr + HALO, lj = pc + HALO;
    float av = s_a[li * DDIM + lj];
    float bv = s_b[li * DDIM + lj];
    float d  = av - bv;
    float w  = (bv > tmean) ? 3.0f : 1.0f;
    float dw = d * w;
    accR += dw * dw;
    float ma = NEGINF, mb = NEGINF;
#pragma unroll
    for (int dy = -1; dy <= 1; ++dy)
#pragma unroll
      for (int dx = -1; dx <= 1; ++dx) {
        ma = fmaxf(ma, s_a[(li + dy) * DDIM + (lj + dx)]);
        mb = fmaxf(mb, s_b[(li + dy) * DDIM + (lj + dx)]);
      }
    accP += fabsf(ma - mb);
  }

  // ---- Reduce: shfl within lane-halves -> ds_add_f32 -> global_atomic_add_f32 ----
#pragma unroll
  for (int e = 0; e < 8; ++e) {
    float v = accD[e];
    v += __shfl_xor(v, 1); v += __shfl_xor(v, 2);
    v += __shfl_xor(v, 4); v += __shfl_xor(v, 8);
    if ((lane & 15) == 0) atomicAdd(&red[e + ((lane >> 4) << 3)], v);
  }
  {
    float v = accR;
    for (int off = 16; off; off >>= 1) v += __shfl_xor(v, off);
    if (lane == 0) atomicAdd(&red[16], v);
    v = accP;
    for (int off = 16; off; off >>= 1) v += __shfl_xor(v, off);
    if (lane == 0) atomicAdd(&red[17], v);
  }
  __syncthreads();
  if (tid < 18) atomicAdd(&ws[tid], red[tid]);
}

// ---------------- Kernel C: combine partial sums into the scalar loss ----------
__global__ void astro_finalize_kernel(const float* __restrict__ ws, float* __restrict__ out) {
  if (threadIdx.x == 0 && blockIdx.x == 0) {
    const float inv = 1.0f / NTOT;
    float conv = 2.0f * (ws[0] + ws[1])              // gradient (sobel x,y)
               + 1.5f * ws[2]                        // point (laplacian)
               + (ws[3] + ws[4] + ws[5])             // gaussian structure
               + (ws[6] + ws[7] + ws[8]);            // LoG scales
    out[0] = (ws[16] + conv + 2.0f * ws[17]) * inv;
  }
}

extern "C" void kernel_launch(void* const* d_in, const int* in_sizes, int n_in,
                              void* d_out, int out_size, void* d_ws, size_t ws_size,
                              hipStream_t stream) {
  const float* rec = (const float*)d_in[0];
  const float* tgt = (const float*)d_in[1];
  float* ws  = (float*)d_ws;
  float* out = (float*)d_out;

  hipMemsetAsync(d_ws, 0, 19 * sizeof(float), stream);

  astro_tsum_kernel<<<1024, 256, 0, stream>>>(tgt, ws, (NIMG * IMG * IMG) / 4);

  dim3 grid(IMG / TILE, IMG / TILE, NIMG);   // 8 x 8 x 64 tiles
  astro_loss_main<<<grid, 256, 0, stream>>>(rec, tgt, ws);

  astro_finalize_kernel<<<1, 32, 0, stream>>>(ws, out);
}
